// Model_19301583028953
// MI455X (gfx1250) — matrix operand order repaired and checked by reading, not compiled, not run on hardware
//
#include <hip/hip_runtime.h>
#include <hip/hip_bf16.h>

typedef __attribute__((ext_vector_type(16))) _Float16 v16h;
typedef __attribute__((ext_vector_type(8)))  float    v8f;

#define DEV_ static __device__ __forceinline__

constexpr int T = 65536, L = 16, CE = 15, WE = 15, H = 30, NTAG = 45, PADC = 84;
constexpr int KX = 32;   // K pad for x (30 -> 32): one 16x16x32 WMMA K-step
constexpr int NG = 128;  // gate count pad (120 -> 128): 8 N-tiles of 16
constexpr int KO = 64;   // K pad for output features (60 -> 64): 2 WMMA K-steps
constexpr int NT = 48;   // tag pad (45 -> 48): 3 N-tiles of 16

// ---- workspace layout (all buffers fully initialized before use) ----
constexpr size_t XH_OFF  = 0;
constexpr size_t XH_BYTES = (size_t)T * KX * 2;          // x_h  [T][32] f16
constexpr size_t G_OFF   = XH_OFF + XH_BYTES;
constexpr size_t G_BYTES  = (size_t)2 * T * NG * 2;      // G    [2][T][128] f16 (precomputed input gates)
constexpr size_t OH_OFF  = G_OFF + G_BYTES;
constexpr size_t OH_BYTES = (size_t)T * KO * 2;          // out_h[T][64] f16 (hf | hb | zero pad)
constexpr size_t WP_OFF  = OH_OFF + OH_BYTES;
constexpr size_t WP_BYTES = (size_t)2 * NG * KX * 2;     // Wih packed f16 [2][128][32]
constexpr size_t BI_OFF  = WP_OFF + WP_BYTES;
constexpr size_t BI_BYTES = (size_t)2 * NG * 4;          // bih+bhh combined f32 [2][128]
constexpr size_t LP_OFF  = BI_OFF + BI_BYTES;
constexpr size_t LP_BYTES = (size_t)NT * KO * 2;         // lin_w packed f16 [48][64]

// ======================= kernel 1: weight packing =======================
__global__ void prep_weights(const float* wih_f, const float* bih_f, const float* bhh_f,
                             const float* wih_b, const float* bih_b, const float* bhh_b,
                             const float* lin_w,
                             _Float16* Wp, float* bias, _Float16* Lp) {
  for (int i = threadIdx.x; i < 2 * NG * KX; i += blockDim.x) {
    int d = i / (NG * KX), rk = i % (NG * KX), r = rk / KX, k = rk % KX;
    const float* w = d ? wih_b : wih_f;
    Wp[i] = (_Float16)((r < 4 * H && k < 2 * WE) ? w[r * (2 * WE) + k] : 0.f);
  }
  for (int i = threadIdx.x; i < 2 * NG; i += blockDim.x) {
    int d = i / NG, r = i % NG;
    const float* b1 = d ? bih_b : bih_f;
    const float* b2 = d ? bhh_b : bhh_f;
    bias[i] = (r < 4 * H) ? (b1[r] + b2[r]) : 0.f;
  }
  for (int i = threadIdx.x; i < NT * KO; i += blockDim.x) {
    int r = i / KO, k = i % KO;
    Lp[i] = (_Float16)((r < NTAG && k < 2 * H) ? lin_w[r * (2 * H) + k] : 0.f);
  }
}

// ============ kernel 2: char conv + maxpool + embed gather -> x_h ============
__global__ void __launch_bounds__(128) build_x(
    const int* word_ids, const int* char_ids, const int* word_lens,
    const float* wordEmbeds, const float* charEmbeds,
    const float* conv_w, const float* conv_b,
    _Float16* xh, _Float16* oh) {
  __shared__ float sCE[85 * CE];
  __shared__ float sCW[CE * 3 * CE];
  __shared__ float sCB[CE];
  for (int i = threadIdx.x; i < 85 * CE; i += blockDim.x) sCE[i] = charEmbeds[i];
  for (int i = threadIdx.x; i < CE * 3 * CE; i += blockDim.x) sCW[i] = conv_w[i];
  for (int i = threadIdx.x; i < CE; i += blockDim.x) sCB[i] = conv_b[i];
  __syncthreads();

  int t = blockIdx.x * blockDim.x + threadIdx.x;
  if (t >= T) return;
  int len = word_lens[t];
  const int* cid = char_ids + (size_t)t * L;

  float cf[CE];
#pragma unroll
  for (int f = 0; f < CE; ++f) cf[f] = -1e30f;

  for (int l = 0; l < L; ++l) {
    if (l >= len) break;                   // maxpool only over valid positions
    float win[3 * CE];
#pragma unroll
    for (int j = 0; j < 3; ++j) {
      int idx = l + j - 1;
      // reference quirk: neighbor valid iff 0 < idx < len (strict), else PAD char
      int c = (idx > 0 && idx < len) ? cid[idx] : PADC;
      const float* e = &sCE[c * CE];
#pragma unroll
      for (int d = 0; d < CE; ++d) win[j * CE + d] = e[d];
    }
#pragma unroll
    for (int f = 0; f < CE; ++f) {
      float s = sCB[f];
      const float* w = &sCW[f * 3 * CE];
#pragma unroll
      for (int k = 0; k < 3 * CE; ++k) s += win[k] * w[k];
      cf[f] = fmaxf(cf[f], s);
    }
  }

  const float* we = wordEmbeds + (size_t)word_ids[t] * WE;
  _Float16* xr = xh + (size_t)t * KX;
#pragma unroll
  for (int k = 0; k < WE; ++k) xr[k] = (_Float16)we[k];
#pragma unroll
  for (int f = 0; f < CE; ++f) xr[WE + f] = (_Float16)cf[f];
  xr[30] = (_Float16)0.f; xr[31] = (_Float16)0.f;

  _Float16* orow = oh + (size_t)t * KO;    // zero out_h (scan fills [0,60))
#pragma unroll
  for (int k = 0; k < KO; ++k) orow[k] = (_Float16)0.f;
}

// ================= WMMA fragment loaders (contiguous K order, same for A and B) =================
// A 16x32 f16: lanes 0-15: M=lane, K 0..15; lanes 16-31: K 16..31 (matches B)
DEV_ v16h load_A16(const _Float16* W, int row0, int rowStride, int kOff, int lane) {
  const _Float16* row = W + (size_t)(row0 + (lane & 15)) * rowStride + kOff;
  int k0 = (lane < 16) ? 0 : 16;
  int k1 = k0 + 8;
  v16h a;
#pragma unroll
  for (int i = 0; i < 8; ++i) { a[i] = row[k0 + i]; a[8 + i] = row[k1 + i]; }
  return a;
}
// B 32x16 f16: lane holds column N=lane&15; halves = 16 consecutive K
// (K 0..15 for lanes 0-15, K 16..31 for lanes 16-31) -> contiguous 32B load
DEV_ v16h load_B16(const _Float16* X, int col0, int colStride, int kOff, int lane) {
  const _Float16* p = X + (size_t)(col0 + (lane & 15)) * colStride + kOff +
                      ((lane < 16) ? 0 : 16);
  v16h b;
#pragma unroll
  for (int i = 0; i < 16; ++i) b[i] = p[i];
  return b;
}

// ====== kernel 3: G[d][t][gate] = x @ Wih_d.T + bih + bhh  (WMMA GEMM) ======
// Computed transposed: D = Wih_pad(A, 16x32) x x^T(B, 32x16) per gate/token tile.
__global__ void __launch_bounds__(256) gate_gemm(const _Float16* xh, const _Float16* Wp,
                                                 const float* bias, _Float16* G) {
  int lane = threadIdx.x & 31;
  int wg = blockIdx.x * (blockDim.x >> 5) + (threadIdx.x >> 5);
  int dir = wg & 1;
  int tile = wg >> 1;
  int t0 = tile * 16;

  v16h b = load_B16(xh, t0, KX, 0, lane);               // 16 tokens, K=0..31
  const _Float16* W = Wp + (size_t)dir * NG * KX;
  const float* bi = bias + dir * NG;
  int t = t0 + (lane & 15);
  int mhalf = (lane < 16) ? 0 : 8;
  _Float16* Grow = G + ((size_t)dir * T + t) * NG;

#pragma unroll
  for (int gt = 0; gt < 8; ++gt) {                      // 8 gate tiles (128 gates)
    v16h a = load_A16(W, gt * 16, KX, 0, lane);
    v8f c = {};
    c = __builtin_amdgcn_wmma_f32_16x16x32_f16(false, a, false, b, (short)0, c, false, false);
    int mb = gt * 16 + mhalf;
    _Float16 out8[8] __attribute__((aligned(16)));
#pragma unroll
    for (int v = 0; v < 8; ++v) out8[v] = (_Float16)(c[v] + bi[mb + v]);
    *reinterpret_cast<uint4*>(Grow + mb) = *reinterpret_cast<const uint4*>(out8);  // 16B store
  }
}

// ============ kernel 4: the serial part — one wave per direction ============
// Lane j owns hidden unit j: all 4 Whh rows in VGPRs, h broadcast via __shfl,
// no barriers; next step's G row double-buffered ahead of the dependence chain.
__global__ void __launch_bounds__(32) lstm_scan(const _Float16* G,
                                                const float* whh_f, const float* whh_b,
                                                _Float16* oh) {
  const int d = blockIdx.x;
  const int j = threadIdx.x;
  const float* whh = d ? whh_b : whh_f;
  const int r = (j < H) ? j : 0;

  float wi[H], wf[H], wg[H], wo[H];
#pragma unroll
  for (int k = 0; k < H; ++k) {
    wi[k] = whh[(r +     0) * H + k];
    wf[k] = whh[(r +     H) * H + k];
    wg[k] = whh[(r + 2 * H) * H + k];
    wo[k] = whh[(r + 3 * H) * H + k];
  }

  const _Float16* Gd = G + (size_t)d * T * NG;
  int t = d ? (T - 1) : 0;
  const int dt = d ? -1 : 1;

  float h = 0.f, c = 0.f;
  const _Float16* Gr0 = Gd + (size_t)t * NG;
  float gi = (float)Gr0[r], gf = (float)Gr0[H + r],
        gg = (float)Gr0[2 * H + r], go = (float)Gr0[3 * H + r];

  for (int s = 0; s < T; ++s) {
    // prefetch next step's gate row before the dependent shuffle chain
    int tn = t + dt;
    float ni = 0.f, nf = 0.f, ng = 0.f, no = 0.f;
    if (s + 1 < T) {
      const _Float16* Gr = Gd + (size_t)tn * NG;
      ni = (float)Gr[r]; nf = (float)Gr[H + r];
      ng = (float)Gr[2 * H + r]; no = (float)Gr[3 * H + r];
    }

    float ai = gi, af = gf, ag = gg, ao = go;
#pragma unroll
    for (int k = 0; k < H; ++k) {
      float hk = __shfl(h, k, 32);
      ai += hk * wi[k]; af += hk * wf[k]; ag += hk * wg[k]; ao += hk * wo[k];
    }
    float sf = 1.f / (1.f + __expf(-af));
    float si = 1.f / (1.f + __expf(-ai));
    float so = 1.f / (1.f + __expf(-ao));
    c = sf * c + si * tanhf(ag);
    h = so * tanhf(c);
    if (j < H) oh[(size_t)t * KO + (d ? H : 0) + j] = (_Float16)h;

    gi = ni; gf = nf; gg = ng; go = no;
    t = tn;
  }
}

// ======= kernel 5: logits = out @ lin_w.T + b, fused softmax (WMMA) =======
__global__ void __launch_bounds__(256) linear_softmax(const _Float16* oh, const _Float16* Lp,
                                                      const float* lin_b, float* out) {
  __shared__ float tile[8][16][NT];
  int lane = threadIdx.x & 31;
  int w = threadIdx.x >> 5;
  int t0 = (blockIdx.x * 8 + w) * 16;

  v16h b0 = load_B16(oh, t0, KO, 0, lane);    // K = 0..31
  v16h b1 = load_B16(oh, t0, KO, 32, lane);   // K = 32..63
  int n = lane & 15;
  int mhalf = (lane < 16) ? 0 : 8;

#pragma unroll
  for (int nt = 0; nt < 3; ++nt) {            // 48 tag rows
    v16h a0 = load_A16(Lp, nt * 16, KO, 0, lane);
    v16h a1 = load_A16(Lp, nt * 16, KO, 32, lane);
    v8f c = {};
    c = __builtin_amdgcn_wmma_f32_16x16x32_f16(false, a0, false, b0, (short)0, c, false, false);
    c = __builtin_amdgcn_wmma_f32_16x16x32_f16(false, a1, false, b1, (short)0, c, false, false);
#pragma unroll
    for (int v = 0; v < 8; ++v) tile[w][n][nt * 16 + mhalf + v] = c[v];
  }
  __syncthreads();

  if (lane < 16) {
    int t = t0 + lane;
    const float* row = tile[w][lane];
    float lg[NTAG];
    float m = -1e30f;
#pragma unroll
    for (int i = 0; i < NTAG; ++i) { lg[i] = row[i] + lin_b[i]; m = fmaxf(m, lg[i]); }
    float s = 0.f;
#pragma unroll
    for (int i = 0; i < NTAG; ++i) { lg[i] = __expf(lg[i] - m); s += lg[i]; }
    float inv = 1.f / s;
    float* o = out + (size_t)t * NTAG;
#pragma unroll
    for (int i = 0; i < NTAG; ++i) o[i] = lg[i] * inv;
  }
}

// ================================ launcher ================================
extern "C" void kernel_launch(void* const* d_in, const int* in_sizes, int n_in,
                              void* d_out, int out_size, void* d_ws, size_t ws_size,
                              hipStream_t stream) {
  const int*   word_ids   = (const int*)d_in[0];
  const int*   char_ids   = (const int*)d_in[1];
  const int*   word_lens  = (const int*)d_in[2];
  const float* wordEmbeds = (const float*)d_in[3];
  const float* charEmbeds = (const float*)d_in[4];
  const float* conv_w     = (const float*)d_in[5];
  const float* conv_b     = (const float*)d_in[6];
  const float* wih_f      = (const float*)d_in[7];
  const float* whh_f      = (const float*)d_in[8];
  const float* bih_f      = (const float*)d_in[9];
  const float* bhh_f      = (const float*)d_in[10];
  const float* wih_b      = (const float*)d_in[11];
  const float* whh_b      = (const float*)d_in[12];
  const float* bih_b      = (const float*)d_in[13];
  const float* bhh_b      = (const float*)d_in[14];
  const float* lin_w      = (const float*)d_in[15];
  const float* lin_b      = (const float*)d_in[16];

  char* ws = (char*)d_ws;
  _Float16* xh   = (_Float16*)(ws + XH_OFF);
  _Float16* G    = (_Float16*)(ws + G_OFF);
  _Float16* oh   = (_Float16*)(ws + OH_OFF);
  _Float16* Wp   = (_Float16*)(ws + WP_OFF);
  float*    bias = (float*)   (ws + BI_OFF);
  _Float16* Lp   = (_Float16*)(ws + LP_OFF);

  prep_weights<<<1, 256, 0, stream>>>(wih_f, bih_f, bhh_f, wih_b, bih_b, bhh_b,
                                      lin_w, Wp, bias, Lp);
  build_x<<<T / 128, 128, 0, stream>>>(word_ids, char_ids, word_lens, wordEmbeds,
                                       charEmbeds, conv_w, conv_b, xh, oh);
  gate_gemm<<<((T / 16) * 2) / 8, 256, 0, stream>>>(xh, Wp, bias, G);
  lstm_scan<<<2, 32, 0, stream>>>(G, whh_f, whh_b, oh);
  linear_softmax<<<(T / 16) / 8, 256, 0, stream>>>(oh, Lp, lin_b, (float*)d_out);
}
